// DiscreteNet_22677427323531
// MI455X (gfx1250) — compile-verified, matrix operand order checked by reading, not run on hardware
//
#include <hip/hip_runtime.h>

typedef __attribute__((ext_vector_type(16))) _Float16 v16h;
typedef __attribute__((ext_vector_type(8)))  _Float16 v8h;
typedef __attribute__((ext_vector_type(8)))  float    v8f;

#define KSTATES 4096
#define ZFK     128      // NV*ED
#define C1      64       // 4*D
#define C2      32       // 2*D
#define C3      16       // D
#define HWPAD   224      // 196 padded to 7*32 for K blocking

// ---- fragment helpers -------------------------------------------------------
// 16-bit A/B lane layout (ISA 7.12.2): lane l (=tid&15), half h (=tid>>4):
// element j (0..15) holds K = (j<8 ? 8*h + j : 16 + 8*h + (j-8)) within a
// 32-wide K block -> two contiguous 8-half (16B) chunks per lane.
static __device__ __forceinline__ v16h frag_load(const _Float16* row, int half) {
    union { v16h v; v8h h[2]; } u;
    u.h[0] = *(const v8h*)(row + 8 * half);
    u.h[1] = *(const v8h*)(row + 16 + 8 * half);
    return u.v;
}

static __device__ __forceinline__ v8f wmma16(v16h a, v16h b, v8f c) {
    return __builtin_amdgcn_wmma_f32_16x16x32_f16(
        /*neg_a=*/false, a, /*neg_b=*/false, b,
        /*c_mod=*/(short)0, c, /*reuse_a=*/false, /*reuse_b=*/false);
}

// ---- prep kernels -----------------------------------------------------------
__global__ void k_prep_zf(const float* __restrict__ le, _Float16* __restrict__ zf) {
    int t = blockIdx.x * blockDim.x + threadIdx.x;
    if (t >= KSTATES * ZFK) return;
    int k = t >> 7, c = t & 127;
    int j = c >> 5, e = c & 31;
    int idx = (k >> (3 * (3 - j))) & 7;
    zf[t] = (_Float16)le[(j * 8 + idx) * 32 + e];
}

__global__ void k_prep_linwT(const float* __restrict__ lw, _Float16* __restrict__ lwT) {
    int t = blockIdx.x * blockDim.x + threadIdx.x;     // 3*4096*128
    if (t >= 3 * KSTATES * ZFK) return;
    int kk = t & 127;
    int n  = (t >> 7) & 4095;
    int v  = t >> 19;
    lwT[t] = (_Float16)lw[((size_t)(v * 128 + kk)) * 4096 + n];
}

__global__ void k_prep_wB1(const float* __restrict__ w1, _Float16* __restrict__ wB1) {
    int t = blockIdx.x * blockDim.x + threadIdx.x;     // 3*9*32*64
    if (t >= 3 * 9 * 32 * 64) return;
    int ci = t & 63, co = (t >> 6) & 31;
    int rem = t >> 11;               // v*9 + dydx
    int dydx = rem % 9, v = rem / 9;
    int dy = dydx / 3, dx = dydx % 3;
    wB1[t] = (_Float16)w1[(((size_t)(v * 64 + ci) * 32 + co) * 3 + (2 - dy)) * 3 + (2 - dx)];
}

__global__ void k_prep_wB2(const float* __restrict__ w2, _Float16* __restrict__ wB2) {
    int t = blockIdx.x * blockDim.x + threadIdx.x;     // 3*9*16*32
    if (t >= 3 * 9 * 16 * 32) return;
    int ci = t & 31, co = (t >> 5) & 15;
    int rem = t >> 9;
    int dydx = rem % 9, v = rem / 9;
    int dy = dydx / 3, dx = dydx % 3;
    wB2[t] = (_Float16)w2[(((size_t)(v * 32 + ci) * 16 + co) * 3 + (2 - dy)) * 3 + (2 - dx)];
}

__global__ void k_prep_wf3(const float* __restrict__ w3, float* __restrict__ wf3) {
    int t = blockIdx.x * blockDim.x + threadIdx.x;     // 3*16*9
    if (t >= 3 * 16 * 9) return;
    int dydx = t % 9;
    int vci  = t / 9;                // v*16 + ci
    int dy = dydx / 3, dx = dydx % 3;
    wf3[t] = w3[(size_t)vci * 9 + (2 - dy) * 3 + (2 - dx)];
}

__global__ void k_prep_x(const float* __restrict__ x, _Float16* __restrict__ xT,
                         float* __restrict__ x2) {
    int t = blockIdx.x * blockDim.x + threadIdx.x;     // 64*3
    if (t >= 64 * 3) return;
    int v = t % 3, b = t / 3;
    const float* src = x + (size_t)(b * 3 + v) * 196;
    _Float16* dst = xT + (size_t)(v * 64 + b) * HWPAD;
    float s = 0.f;
    for (int hw = 0; hw < 196; ++hw) {
        float val = src[hw];
        dst[hw] = (_Float16)val;
        s += val * val;
    }
    for (int hw = 196; hw < HWPAD; ++hw) dst[hw] = (_Float16)0.f;
    x2[b * 3 + v] = s;
}

// ---- linear: H1[pix][k][ci] = zf @ lin_w + lin_b (reshaped) -----------------
// Two M-tiles per wave: B fragments loaded once per N tile, used twice.
__global__ void k_linear(const _Float16* __restrict__ zf,
                         const _Float16* __restrict__ linwT,   // view slice [4096][128]
                         const float* __restrict__ lin_b,      // view slice [4096]
                         _Float16* __restrict__ H1) {
    const int lane = threadIdx.x & 31;
    const int l = lane & 15, half = lane >> 4;
    const int Mb0 = blockIdx.y * 2;
    const int Mb1 = Mb0 + 1;
    const int g   = blockIdx.x;
    const _Float16* arow0 = zf + (size_t)(Mb0 * 16 + l) * ZFK;
    const _Float16* arow1 = zf + (size_t)(Mb1 * 16 + l) * ZFK;
    v16h a00 = frag_load(arow0 +  0, half);
    v16h a01 = frag_load(arow0 + 32, half);
    v16h a02 = frag_load(arow0 + 64, half);
    v16h a03 = frag_load(arow0 + 96, half);
    v16h a10 = frag_load(arow1 +  0, half);
    v16h a11 = frag_load(arow1 + 32, half);
    v16h a12 = frag_load(arow1 + 64, half);
    v16h a13 = frag_load(arow1 + 96, half);
    for (int nbi = 0; nbi < 16; ++nbi) {
        const int Nb = g * 16 + nbi;
        const _Float16* brow = linwT + (size_t)(Nb * 16 + l) * ZFK;
        v16h b0 = frag_load(brow +  0, half);
        v16h b1 = frag_load(brow + 32, half);
        v16h b2 = frag_load(brow + 64, half);
        v16h b3 = frag_load(brow + 96, half);
        v8f acc0 = {}, acc1 = {};
        acc0 = wmma16(a00, b0, acc0);  acc1 = wmma16(a10, b0, acc1);
        acc0 = wmma16(a01, b1, acc0);  acc1 = wmma16(a11, b1, acc1);
        acc0 = wmma16(a02, b2, acc0);  acc1 = wmma16(a12, b2, acc1);
        acc0 = wmma16(a03, b3, acc0);  acc1 = wmma16(a13, b3, acc1);
        const int c = Nb * 16 + l;
        const float bias = lin_b[c];
        const int pix = c & 63;        // c = ci*64 + pix  (reshape K,64,8,8)
        const int ci  = c >> 6;
        _Float16* outp = H1 + (size_t)pix * KSTATES * C1 + ci;
#pragma unroll
        for (int r = 0; r < 8; ++r) {
            int k0 = Mb0 * 16 + r + 8 * half;
            int k1 = Mb1 * 16 + r + 8 * half;
            outp[(size_t)k0 * C1] = (_Float16)(acc0[r] + bias);
            outp[(size_t)k1 * C1] = (_Float16)(acc1[r] + bias);
        }
    }
}

// ---- conv1 (implicit GEMM per output pixel), ELU, -> H2[pix][k][co] ---------
// Full N=32 per wave: A fragments loaded once per tap, used for both co halves.
__global__ void k_conv1(const _Float16* __restrict__ H1,
                        const _Float16* __restrict__ wB1,   // view slice [9][32][64]
                        const float* __restrict__ b1,       // view slice [32]
                        _Float16* __restrict__ H2) {
    const int lane = threadIdx.x & 31;
    const int l = lane & 15, half = lane >> 4;
    const int Mb = blockIdx.x;     // 0..255
    const int pix = blockIdx.y;    // 0..99
    const int oy = pix / 10, ox = pix % 10;
    v8f acc0 = {}, acc1 = {};
    for (int dy = 0; dy < 3; ++dy) {
        int iy = oy + dy - 2;
        if (iy < 0 || iy >= 8) continue;
        for (int dx = 0; dx < 3; ++dx) {
            int ix = ox + dx - 2;
            if (ix < 0 || ix >= 8) continue;
            const _Float16* arow  = H1 + ((size_t)(iy * 8 + ix) * KSTATES + Mb * 16 + l) * C1;
            const _Float16* brow0 = wB1 + ((size_t)(dy * 3 + dx) * 32 + l)      * C1;
            const _Float16* brow1 = wB1 + ((size_t)(dy * 3 + dx) * 32 + 16 + l) * C1;
            v16h aA = frag_load(arow,      half);
            v16h aB = frag_load(arow + 32, half);
            acc0 = wmma16(aA, frag_load(brow0,      half), acc0);
            acc0 = wmma16(aB, frag_load(brow0 + 32, half), acc0);
            acc1 = wmma16(aA, frag_load(brow1,      half), acc1);
            acc1 = wmma16(aB, frag_load(brow1 + 32, half), acc1);
        }
    }
    const float bias0 = b1[l];
    const float bias1 = b1[16 + l];
    _Float16* outp = H2 + (size_t)(oy * 10 + ox) * KSTATES * C2;
#pragma unroll
    for (int r = 0; r < 8; ++r) {
        int k = Mb * 16 + r + 8 * half;
        float v0 = acc0[r] + bias0;
        float v1 = acc1[r] + bias1;
        v0 = v0 > 0.f ? v0 : expm1f(v0);
        v1 = v1 > 0.f ? v1 : expm1f(v1);
        outp[(size_t)k * C2 + l]      = (_Float16)v0;
        outp[(size_t)k * C2 + 16 + l] = (_Float16)v1;
    }
}

// ---- conv2, ELU, -> H3[pix][k][co] ------------------------------------------
__global__ void k_conv2(const _Float16* __restrict__ H2,
                        const _Float16* __restrict__ wB2,   // view slice [9][16][32]
                        const float* __restrict__ b2,       // view slice [16]
                        _Float16* __restrict__ H3) {
    const int lane = threadIdx.x & 31;
    const int l = lane & 15, half = lane >> 4;
    const int Mb = blockIdx.x;     // 0..255
    const int pix = blockIdx.y;    // 0..143
    const int oy = pix / 12, ox = pix % 12;
    v8f acc = {};
    for (int dy = 0; dy < 3; ++dy) {
        int iy = oy + dy - 2;
        if (iy < 0 || iy >= 10) continue;
        for (int dx = 0; dx < 3; ++dx) {
            int ix = ox + dx - 2;
            if (ix < 0 || ix >= 10) continue;
            const _Float16* arow = H2 + ((size_t)(iy * 10 + ix) * KSTATES + Mb * 16 + l) * C2;
            const _Float16* brow = wB2 + ((size_t)(dy * 3 + dx) * 16 + l) * C2;
            acc = wmma16(frag_load(arow, half), frag_load(brow, half), acc);
        }
    }
    const int co = l;
    const float bias = b2[co];
    _Float16* outp = H3 + (size_t)(oy * 12 + ox) * KSTATES * C3 + co;
#pragma unroll
    for (int r = 0; r < 8; ++r) {
        int k = Mb * 16 + r + 8 * half;
        float val = acc[r] + bias;
        val = val > 0.f ? val : expm1f(val);
        outp[(size_t)k * C3] = (_Float16)val;
    }
}

// ---- conv3 (16->1), m & m2 --------------------------------------------------
__global__ void k_conv3(const _Float16* __restrict__ H3,
                        const float* __restrict__ wf3,   // view slice [16][9]
                        const float* __restrict__ b3,    // view slice [1]
                        _Float16* __restrict__ mrow,     // view slice [4096][224]
                        float* __restrict__ m2, int v) {
    int k = blockIdx.x * blockDim.x + threadIdx.x;
    if (k >= KSTATES) return;
    const float bias = b3[0];
    float m2loc = 0.f;
    for (int oy = 0; oy < 14; ++oy) {
        for (int ox = 0; ox < 14; ++ox) {
            float acc = bias;
            for (int dy = 0; dy < 3; ++dy) {
                int iy = oy + dy - 2;
                if (iy < 0 || iy >= 12) continue;
                for (int dx = 0; dx < 3; ++dx) {
                    int ix = ox + dx - 2;
                    if (ix < 0 || ix >= 12) continue;
                    const _Float16* p = H3 + ((size_t)(iy * 12 + ix) * KSTATES + k) * C3;
                    const float* w = wf3 + (dy * 3 + dx);
#pragma unroll
                    for (int ci = 0; ci < 16; ++ci)
                        acc += (float)p[ci] * w[ci * 9];
                }
            }
            m2loc += acc * acc;
            mrow[(size_t)k * HWPAD + oy * 14 + ox] = (_Float16)acc;
        }
    }
    for (int hw = 196; hw < HWPAD; ++hw) mrow[(size_t)k * HWPAD + hw] = (_Float16)0.f;
    m2[k * 3 + v] = m2loc;
}

// ---- cross GEMM + fused epilogue: out[b][k][v] = cross - 0.5*(m2+x2) --------
__global__ void k_cross(const _Float16* __restrict__ mrow,  // view slice [4096][224]
                        const _Float16* __restrict__ xT,    // view slice [64][224]
                        const float* __restrict__ m2,
                        const float* __restrict__ x2,
                        float* __restrict__ out, int v) {
    const int lane = threadIdx.x & 31;
    const int l = lane & 15, half = lane >> 4;
    const int Nb = blockIdx.x;     // 0..3 (batch tiles)
    const int Mb = blockIdx.y;     // 0..255 (state tiles)
    const _Float16* arow = mrow + (size_t)(Mb * 16 + l) * HWPAD;
    const _Float16* brow = xT   + (size_t)(Nb * 16 + l) * HWPAD;
    v8f acc = {};
#pragma unroll
    for (int kb = 0; kb < 7; ++kb)
        acc = wmma16(frag_load(arow + kb * 32, half), frag_load(brow + kb * 32, half), acc);
    const int b = Nb * 16 + l;
    const float x2v = x2[b * 3 + v];
#pragma unroll
    for (int r = 0; r < 8; ++r) {
        int k = Mb * 16 + r + 8 * half;
        out[((size_t)b * KSTATES + k) * 3 + v] = acc[r] - 0.5f * (m2[k * 3 + v] + x2v);
    }
}

// ---- host -------------------------------------------------------------------
static size_t align256(size_t x) { return (x + 255) & ~(size_t)255; }

extern "C" void kernel_launch(void* const* d_in, const int* in_sizes, int n_in,
                              void* d_out, int out_size, void* d_ws, size_t ws_size,
                              hipStream_t stream) {
    (void)in_sizes; (void)n_in; (void)out_size; (void)ws_size;
    const float* x_in = (const float*)d_in[0];
    const float* le   = (const float*)d_in[1];
    const float* lw   = (const float*)d_in[2];
    const float* lb   = (const float*)d_in[3];
    const float* w1   = (const float*)d_in[4];
    const float* b1   = (const float*)d_in[5];
    const float* w2   = (const float*)d_in[6];
    const float* b2   = (const float*)d_in[7];
    const float* w3   = (const float*)d_in[8];
    const float* b3   = (const float*)d_in[9];
    float* out = (float*)d_out;

    char* ws = (char*)d_ws;
    size_t off = 0;
    _Float16* zf    = (_Float16*)(ws + off); off = align256(off + (size_t)KSTATES * ZFK * 2);
    _Float16* linwT = (_Float16*)(ws + off); off = align256(off + (size_t)3 * KSTATES * ZFK * 2);
    _Float16* wB1   = (_Float16*)(ws + off); off = align256(off + (size_t)3 * 9 * 32 * 64 * 2);
    _Float16* wB2   = (_Float16*)(ws + off); off = align256(off + (size_t)3 * 9 * 16 * 32 * 2);
    float*    wf3   = (float*)   (ws + off); off = align256(off + (size_t)3 * 16 * 9 * 4);
    _Float16* xT    = (_Float16*)(ws + off); off = align256(off + (size_t)3 * 64 * HWPAD * 2);
    float*    x2    = (float*)   (ws + off); off = align256(off + (size_t)64 * 3 * 4);
    float*    m2    = (float*)   (ws + off); off = align256(off + (size_t)KSTATES * 3 * 4);
    _Float16* mbuf  = (_Float16*)(ws + off); off = align256(off + (size_t)3 * KSTATES * HWPAD * 2);
    _Float16* H1    = (_Float16*)(ws + off); off = align256(off + (size_t)64  * KSTATES * C1 * 2);
    _Float16* H2    = (_Float16*)(ws + off); off = align256(off + (size_t)100 * KSTATES * C2 * 2);
    _Float16* H3    = (_Float16*)(ws + off); off = align256(off + (size_t)144 * KSTATES * C3 * 2);

    // prep
    k_prep_zf   <<<(KSTATES * ZFK + 255) / 256, 256, 0, stream>>>(le, zf);
    k_prep_linwT<<<(3 * KSTATES * ZFK + 255) / 256, 256, 0, stream>>>(lw, linwT);
    k_prep_wB1  <<<(3 * 9 * 32 * 64 + 255) / 256, 256, 0, stream>>>(w1, wB1);
    k_prep_wB2  <<<(3 * 9 * 16 * 32 + 255) / 256, 256, 0, stream>>>(w2, wB2);
    k_prep_wf3  <<<(3 * 16 * 9 + 255) / 256, 256, 0, stream>>>(w3, wf3);
    k_prep_x    <<<1, 256, 0, stream>>>(x_in, xT, x2);

    for (int v = 0; v < 3; ++v) {
        k_linear<<<dim3(16, 128), 32, 0, stream>>>(
            zf, linwT + (size_t)v * KSTATES * ZFK, lb + (size_t)v * 4096, H1);
        k_conv1<<<dim3(256, 100), 32, 0, stream>>>(
            H1, wB1 + (size_t)v * 9 * 32 * 64, b1 + v * 32, H2);
        k_conv2<<<dim3(256, 144), 32, 0, stream>>>(
            H2, wB2 + (size_t)v * 9 * 16 * 32, b2 + v * 16, H3);
        k_conv3<<<16, 256, 0, stream>>>(
            H3, wf3 + (size_t)v * 16 * 9, b3 + v,
            mbuf + (size_t)v * KSTATES * HWPAD, m2, v);
        k_cross<<<dim3(4, 256), 32, 0, stream>>>(
            mbuf + (size_t)v * KSTATES * HWPAD, xT + (size_t)v * 64 * HWPAD,
            m2, x2, out, v);
    }
}
